// MoEAutoencoder_62423054680312
// MI455X (gfx1250) — compile-verified, matrix operand order checked by reading, not run on hardware
//
#include <hip/hip_runtime.h>
#include <hip/hip_bf16.h>
#include <math.h>

typedef __attribute__((ext_vector_type(16))) _Float16 v16h;
typedef __attribute__((ext_vector_type(8)))  _Float16 v8h;
typedef __attribute__((ext_vector_type(4)))  _Float16 v4h;
typedef __attribute__((ext_vector_type(8)))  float    v8f;

#define BN_EPS 1e-5f

// ---------------------------------------------------------------------------
// CDNA5 WMMA f16 16x16x32 fragment helpers (wave32).
// A (16Mx32K): lane L holds row m=(L&15); k-base kb=(L<16)?0:8;
//              half h in [0,16): k = kb + (h<8 ? h : h+8)
// B (32Kx16N): lane L holds col n=(L&15); same K striping.
// C/D (16x16 f32, 8 VGPRs): lane L -> n=(L&15); vgpr r -> m = r + ((L<16)?0:8)
// ---------------------------------------------------------------------------
__device__ __forceinline__ int frag_k(int kb, int h) {
  return kb + ((h < 8) ? h : (h + 8));
}

// Build a B fragment from an LDS-staged [16 col][32 k] f16 tile.
__device__ __forceinline__ v16h load_b_frag(const _Float16* bshRow /*&Bsh[p][col][0]*/, int kb) {
  v8h lo = *(const v8h*)(bshRow + kb);        // k = kb .. kb+7   (16B aligned)
  v8h hi = *(const v8h*)(bshRow + kb + 16);   // k = kb+16 .. kb+23
  v16h f;
#pragma unroll
  for (int i = 0; i < 8; ++i) { f[i] = lo[i]; f[i + 8] = hi[i]; }
  return f;
}

// ===========================================================================
// Conv2d 3x3 s2 p1 as implicit GEMM.  K t-major: k = t*CIN + ci (t = ky*3+kx)
// so for CIN>=32 (pow2) each 32-wide K-step has ONE tap.  Block = 4 waves =
// 4 consecutive m-tiles sharing one n-tile; the 32x16 B (im2col) tile is
// gathered cooperatively into double-buffered LDS once per K-step.
// Weights [Cout,Cin,3,3] -> A[m,k] = w[m*K + ci*9 + t] (A rows m>=Cout are
// address-clamped only: their D rows are never stored).
// ===========================================================================
template<int CIN>
__global__ void conv3x3s2_wmma(const float* __restrict__ x, const float* __restrict__ w,
                               const float* __restrict__ bias, float* __restrict__ y,
                               int B, int Hin, int Win, int Cout, int Ho, int Wo)
{
  __shared__ _Float16 Bsh[2][16][32];
  constexpr int K = CIN * 9;

  const int tid  = threadIdx.x;
  const int lane = tid & 31;
  const int wid  = tid >> 5;
  const int n0   = blockIdx.x * 16;
  const int Npix = B * Ho * Wo;
  if (n0 >= Npix) return;                       // block-uniform

  const int mTiles  = (Cout + 15) >> 4;
  const int mTileRw = blockIdx.y * 4 + wid;
  const int mTile   = mTileRw < mTiles ? mTileRw : mTiles - 1;
  const bool mOK    = (mTileRw < mTiles);
  const int m0      = mTile * 16;

  const int HW  = Hin * Win;
  const int col = lane & 15;
  const int kb  = (lane < 16) ? 0 : 8;
  const int mA  = m0 + col;
  const int mAc = mA < Cout ? mA : Cout - 1;

  // ---- staging role: thread covers column scol, k = skk..skk+3 of the step
  const int scol = tid >> 3;
  const int skk  = (tid & 7) * 4;
  const int sn   = (n0 + scol) < Npix ? (n0 + scol) : (Npix - 1);
  const int spb  = sn / (Ho * Wo);
  const int srr  = sn % (Ho * Wo);
  const int spoy = srr / Wo, spox = srr % Wo;

  v8f acc = {};
  if constexpr (CIN >= 32) {
    for (int k0 = 0; k0 < K; k0 += 32) {
      const int p   = (k0 >> 5) & 1;
      const int t   = k0 / CIN;                 // shift (CIN pow2)
      const int ciB = k0 - t * CIN;
      const int ky = t / 3, kx = t - ky * 3;
      // -- cooperative im2col gather -> LDS (f16) --
      {
        const int iy = spoy * 2 - 1 + ky;
        const int ix = spox * 2 - 1 + kx;
        const bool bm = (iy >= 0) & (iy < Hin) & (ix >= 0) & (ix < Win);
        const float* xb = x + (size_t)(spb * CIN + ciB + skk) * HW
                            + (bm ? (iy * Win + ix) : 0);
        float f0 = xb[0], f1 = xb[HW], f2 = xb[2 * HW], f3 = xb[3 * HW];
        v4h pk;
        pk[0] = (_Float16)(bm ? f0 : 0.f);
        pk[1] = (_Float16)(bm ? f1 : 0.f);
        pk[2] = (_Float16)(bm ? f2 : 0.f);
        pk[3] = (_Float16)(bm ? f3 : 0.f);
        *(v4h*)&Bsh[p][scol][skk] = pk;
      }
      __syncthreads();
      // -- A fragment from global (weights are L2-resident) --
      const float* wb = w + (size_t)mAc * K + ciB * 9 + t;
      v16h afrag;
#pragma unroll
      for (int h = 0; h < 16; ++h)
        afrag[h] = (_Float16)wb[frag_k(kb, h) * 9];
      v16h bfrag = load_b_frag(&Bsh[p][col][0], kb);
      acc = __builtin_amdgcn_wmma_f32_16x16x32_f16(false, afrag, false, bfrag,
                                                   (short)0, acc, false, false);
    }
  } else {
    // ---- generic path (conv1, CIN=3, K=27: one masked K-step) ----
    for (int k0 = 0; k0 < K; k0 += 32) {
      const int p = (k0 >> 5) & 1;
      {
        v4h pk;
#pragma unroll
        for (int j = 0; j < 4; ++j) {
          const int kg = k0 + skk + j;
          const bool kv = (kg < K);
          const int kc = kv ? kg : K - 1;
          const int t  = kc / CIN;
          const int ci = kc - t * CIN;
          const int ky = t / 3, kx = t - ky * 3;
          const int iy = spoy * 2 - 1 + ky;
          const int ix = spox * 2 - 1 + kx;
          const bool bm = kv & (iy >= 0) & (iy < Hin) & (ix >= 0) & (ix < Win);
          float f = x[(size_t)(spb * CIN + ci) * HW + (bm ? (iy * Win + ix) : 0)];
          pk[j] = (_Float16)(bm ? f : 0.f);
        }
        *(v4h*)&Bsh[p][scol][skk] = pk;
      }
      __syncthreads();
      v16h afrag;
#pragma unroll
      for (int h = 0; h < 16; ++h) {
        const int k  = k0 + frag_k(kb, h);
        const bool kv = (k < K);
        const int kc = kv ? k : K - 1;
        const int t  = kc / CIN;
        const int ci = kc - t * CIN;
        float av = w[(size_t)mAc * K + ci * 9 + t];
        afrag[h] = (_Float16)(kv ? av : 0.f);
      }
      v16h bfrag = load_b_frag(&Bsh[p][col][0], kb);
      acc = __builtin_amdgcn_wmma_f32_16x16x32_f16(false, afrag, false, bfrag,
                                                   (short)0, acc, false, false);
    }
  }

  const int nE = n0 + col;
  if (mOK && nE < Npix) {
    const int pb  = nE / (Ho * Wo);
    const int rr  = nE % (Ho * Wo);
    const int poy = rr / Wo, pox = rr % Wo;
#pragma unroll
    for (int r = 0; r < 8; ++r) {
      const int m = m0 + r + ((lane < 16) ? 0 : 8);
      if (m < Cout)
        y[((size_t)(pb * Cout + m) * Ho + poy) * Wo + pox] = acc[r] + bias[m];
    }
  }
}

// ===========================================================================
// ConvTranspose2d k4 s2 p1, sub-pixel decomposition (gridDim.z = 4 parities).
// Virtual concat: CIN = C1+C2, C1 = C2 = CIN/2 (pow2 >= 32).  K = CIN*4,
// t-major: one (ky,kx) tap and one source tensor per 32-wide K-step.
// Same block-cooperative LDS staging as the conv kernel.
// Torch weight layout [Cin, Cout, 4, 4].  Optional fused sigmoid.
// ===========================================================================
template<int CIN>
__global__ void deconv4x4s2_wmma(const float* __restrict__ in1,
                                 const float* __restrict__ in2,
                                 const float* __restrict__ w, const float* __restrict__ bias,
                                 float* __restrict__ y,
                                 int B, int Hin, int Win, int Cout, int Ho, int Wo,
                                 int applySigmoid)
{
  __shared__ _Float16 Bsh[2][16][32];
  constexpr int C1 = CIN / 2;
  constexpr int K  = CIN * 4;          // multiple of 32

  const int tid  = threadIdx.x;
  const int lane = tid & 31;
  const int wid  = tid >> 5;
  const int py   = (blockIdx.z >> 1) & 1;
  const int px   = blockIdx.z & 1;

  const int Hop = Ho >> 1, Wop = Wo >> 1;
  const int Npar = B * Hop * Wop;
  const int n0 = blockIdx.x * 16;
  if (n0 >= Npar) return;              // block-uniform

  const int mTiles  = (Cout + 15) >> 4;
  const int mTileRw = blockIdx.y * 4 + wid;
  const int mTile   = mTileRw < mTiles ? mTileRw : mTiles - 1;
  const bool mOK    = (mTileRw < mTiles);
  const int m0      = mTile * 16;

  const int HW  = Hin * Win;
  const int col = lane & 31 & 15;
  const int kb  = (lane < 16) ? 0 : 8;
  const int pym = (py + 1) & 1;
  const int pxm = (px + 1) & 1;
  const int mA  = m0 + col;
  const int mAc = mA < Cout ? mA : Cout - 1;

  // staging role
  const int scol = tid >> 3;
  const int skk  = (tid & 7) * 4;
  const int sn   = (n0 + scol) < Npar ? (n0 + scol) : (Npar - 1);
  const int spb  = sn / (Hop * Wop);
  const int srr  = sn % (Hop * Wop);
  const int soy  = (srr / Wop) * 2 + py;
  const int sox  = (srr % Wop) * 2 + px;

  v8f acc = {};
  for (int k0 = 0; k0 < K; k0 += 32) {
    const int p   = (k0 >> 5) & 1;
    const int t   = k0 / CIN;                 // shift (CIN pow2)
    const int ciB = k0 - t * CIN;             // step stays inside one source tensor
    const int ky = pym + 2 * (t >> 1);
    const int kx = pxm + 2 * (t & 1);
    // -- cooperative gather -> LDS --
    {
      const int iy = (soy + 1 - ky) / 2;
      const int ix = (sox + 1 - kx) / 2;
      const bool bm = (iy >= 0) & (iy < Hin) & (ix >= 0) & (ix < Win);
      const float* src = (ciB < C1) ? in1 : in2;
      const int cc0 = (ciB < C1) ? ciB : (ciB - C1);
      const float* xb = src + (size_t)(spb * C1 + cc0 + skk) * HW
                            + (bm ? (iy * Win + ix) : 0);
      float f0 = xb[0], f1 = xb[HW], f2 = xb[2 * HW], f3 = xb[3 * HW];
      v4h pk;
      pk[0] = (_Float16)(bm ? f0 : 0.f);
      pk[1] = (_Float16)(bm ? f1 : 0.f);
      pk[2] = (_Float16)(bm ? f2 : 0.f);
      pk[3] = (_Float16)(bm ? f3 : 0.f);
      *(v4h*)&Bsh[p][scol][skk] = pk;
    }
    __syncthreads();
    // -- A fragment from global --
    const float* wb = w + (size_t)ciB * Cout * 16 + (size_t)mAc * 16 + ky * 4 + kx;
    v16h afrag;
#pragma unroll
    for (int h = 0; h < 16; ++h)
      afrag[h] = (_Float16)wb[(size_t)frag_k(kb, h) * Cout * 16];
    v16h bfrag = load_b_frag(&Bsh[p][col][0], kb);
    acc = __builtin_amdgcn_wmma_f32_16x16x32_f16(false, afrag, false, bfrag,
                                                 (short)0, acc, false, false);
  }

  const int nE = n0 + col;
  if (mOK && nE < Npar) {
    const int pb = nE / (Hop * Wop);
    const int rr = nE % (Hop * Wop);
    const int oy = (rr / Wop) * 2 + py;
    const int ox = (rr % Wop) * 2 + px;
#pragma unroll
    for (int r = 0; r < 8; ++r) {
      const int m = m0 + r + ((lane < 16) ? 0 : 8);
      if (m < Cout) {
        float v = acc[r] + bias[m];
        if (applySigmoid) v = 1.0f / (1.0f + expf(-v));
        y[((size_t)(pb * Cout + m) * Ho + oy) * Wo + ox] = v;
      }
    }
  }
}

// ===========================================================================
// GEMM, both operands "transposed" view (all FC call sites):
//   C[m,n] = sum_k A[k*lda + m] * B[n*ldb + k] + bias[m],  K % 32 == 0.
// B contiguous in k per lane -> float4 (global_load_b128).
// ===========================================================================
__global__ void gemm_tt_wmma(const float* __restrict__ A, int lda,
                             const float* __restrict__ Bm, int ldb,
                             const float* __restrict__ bias,
                             float* __restrict__ out, int osm, int osn,
                             int M, int N, int K)
{
  const int lane  = threadIdx.x & 31;
  const int wid   = threadIdx.x >> 5;
  const int nTile = blockIdx.x * (blockDim.x >> 5) + wid;
  const int m0    = blockIdx.y * 16;
  const int n0    = nTile * 16;
  if (n0 >= N) return;

  const int col = lane & 15;
  const int kb  = (lane < 16) ? 0 : 8;
  const int mA  = m0 + col;
  const int mAc = mA < M ? mA : M - 1;
  const int nB  = n0 + col;
  const int nBc = nB < N ? nB : N - 1;
  const float* brow = Bm + (size_t)nBc * ldb;

  v8f acc = {};
  for (int k0 = 0; k0 < K; k0 += 32) {
    const float*  ab = A + (size_t)(k0 + kb) * lda + mAc;
    const float4* bp = (const float4*)(brow + k0 + kb);   // 16B aligned at all call sites
    float4 b0 = bp[0], b1 = bp[1], b2 = bp[4], b3 = bp[5];
    v16h afrag, bfrag;
#pragma unroll
    for (int h = 0; h < 8; ++h)
      afrag[h] = (_Float16)ab[(size_t)h * lda];
#pragma unroll
    for (int h = 8; h < 16; ++h)
      afrag[h] = (_Float16)ab[(size_t)(h + 8) * lda];
    bfrag[0]=(_Float16)b0.x; bfrag[1]=(_Float16)b0.y; bfrag[2]=(_Float16)b0.z; bfrag[3]=(_Float16)b0.w;
    bfrag[4]=(_Float16)b1.x; bfrag[5]=(_Float16)b1.y; bfrag[6]=(_Float16)b1.z; bfrag[7]=(_Float16)b1.w;
    bfrag[8]=(_Float16)b2.x; bfrag[9]=(_Float16)b2.y; bfrag[10]=(_Float16)b2.z; bfrag[11]=(_Float16)b2.w;
    bfrag[12]=(_Float16)b3.x; bfrag[13]=(_Float16)b3.y; bfrag[14]=(_Float16)b3.z; bfrag[15]=(_Float16)b3.w;
    acc = __builtin_amdgcn_wmma_f32_16x16x32_f16(false, afrag, false, bfrag,
                                                 (short)0, acc, false, false);
  }

  if (nB < N) {
#pragma unroll
    for (int r = 0; r < 8; ++r) {
      const int m = m0 + r + ((lane < 16) ? 0 : 8);
      if (m < M)
        out[(size_t)m * osm + (size_t)nB * osn] = acc[r] + bias[m];
    }
  }
}

// ==========================================================
// BatchNorm training stats -> per-channel scale/shift
// ==========================================================
__global__ void bn_stats(const float* __restrict__ y, const float* __restrict__ g,
                         const float* __restrict__ bsh, float* __restrict__ scale_shift,
                         int B, int C, int HW)
{
  __shared__ float ssum[256];
  __shared__ float ssq[256];
  const int c = blockIdx.x;
  const int tid = threadIdx.x;
  const int total = B * HW;
  float s = 0.f, q = 0.f;
  for (int t = tid; t < total; t += 256) {
    int b = t / HW, i = t - b * HW;
    float v = y[((size_t)b * C + c) * HW + i];
    s += v; q += v * v;
  }
  ssum[tid] = s; ssq[tid] = q;
  __syncthreads();
  for (int d = 128; d > 0; d >>= 1) {
    if (tid < d) { ssum[tid] += ssum[tid + d]; ssq[tid] += ssq[tid + d]; }
    __syncthreads();
  }
  if (tid == 0) {
    float inv = 1.0f / (float)total;
    float mean = ssum[0] * inv;
    float var  = ssq[0] * inv - mean * mean;
    float sc = g[c] * rsqrtf(var + BN_EPS);
    scale_shift[2 * c]     = sc;
    scale_shift[2 * c + 1] = bsh[c] - mean * sc;
  }
}

__global__ void bn_apply_relu(const float* __restrict__ y, const float* __restrict__ scale_shift,
                              float* __restrict__ dst, int C, int HW, long total)
{
  long i = (long)blockIdx.x * blockDim.x + threadIdx.x;
  if (i >= total) return;
  int c = (int)((i / HW) % C);
  float v = y[i] * scale_shift[2 * c] + scale_shift[2 * c + 1];
  dst[i] = v > 0.f ? v : 0.f;
}

// ==========================================================
// Softmax + top-2 (one thread per row), E = 64
// ==========================================================
__global__ void softmax_top2(const float* __restrict__ logits, float* __restrict__ gate,
                             int* __restrict__ topidx, float* __restrict__ topw,
                             int Bn, int E)
{
  int b = blockIdx.x * blockDim.x + threadIdx.x;
  if (b >= Bn) return;
  const float* row = logits + (size_t)b * E;
  float mx = row[0];
  for (int e = 1; e < E; ++e) mx = fmaxf(mx, row[e]);
  float sum = 0.f;
  for (int e = 0; e < E; ++e) sum += expf(row[e] - mx);
  float inv = 1.0f / sum;
  float v1 = -1.f, v2 = -1.f; int i1 = 0, i2 = 0;
  for (int e = 0; e < E; ++e) {
    float gv = expf(row[e] - mx) * inv;
    gate[(size_t)b * E + e] = gv;
    if (gv > v1)      { v2 = v1; i2 = i1; v1 = gv; i1 = e; }
    else if (gv > v2) { v2 = gv; i2 = e; }
  }
  float wsum = v1 + v2;
  topidx[b * 2]     = i1;
  topidx[b * 2 + 1] = i2;
  topw[b * 2]       = v1 / wsum;
  topw[b * 2 + 1]   = v2 / wsum;
}

// ==========================================================
// Sparse top-2 residual experts (one 64-thread block / sample)
// ==========================================================
__global__ void expert_apply(const float* __restrict__ z,
                             const float* __restrict__ w1, const float* __restrict__ b1,
                             const float* __restrict__ w2, const float* __restrict__ b2,
                             const int* __restrict__ topidx, const float* __restrict__ topw,
                             float* __restrict__ zout, int L, int H)
{
  __shared__ float zsh[64];
  __shared__ float hsh[64];
  const int b = blockIdx.x;
  const int j = threadIdx.x;
  zsh[j] = z[(size_t)b * L + j];
  __syncthreads();
  float accum = zsh[j];   // top-2 weights renormalized to sum 1
  for (int k = 0; k < 2; ++k) {
    int e = topidx[b * 2 + k];
    const float* W1 = w1 + (size_t)e * L * H;
    const float* W2 = w2 + (size_t)e * H * L;
    float hv = b1[(size_t)e * H + j];
    for (int l = 0; l < 64; ++l) hv += zsh[l] * W1[l * H + j];
    hv = hv > 0.f ? hv : 0.f;
    hsh[j] = hv;
    __syncthreads();
    float ov = b2[(size_t)e * L + j];
    for (int hh = 0; hh < 64; ++hh) ov += hsh[hh] * W2[hh * L + j];
    accum += topw[b * 2 + k] * ov;
    __syncthreads();
  }
  zout[(size_t)b * L + j] = accum;
}

// ==========================================================================
// Host-side orchestration
// ==========================================================================
static inline int cdiv(int a, int b) { return (a + b - 1) / b; }

extern "C" void kernel_launch(void* const* d_in, const int* in_sizes, int n_in,
                              void* d_out, int out_size, void* d_ws, size_t ws_size,
                              hipStream_t stream) {
  (void)in_sizes; (void)n_in; (void)out_size; (void)ws_size;

  const float* x     = (const float*)d_in[0];
  const float* c1_w  = (const float*)d_in[1];  const float* c1_b  = (const float*)d_in[2];
  const float* bn1_g = (const float*)d_in[3];  const float* bn1_b = (const float*)d_in[4];
  const float* c2_w  = (const float*)d_in[5];  const float* c2_b  = (const float*)d_in[6];
  const float* bn2_g = (const float*)d_in[7];  const float* bn2_b = (const float*)d_in[8];
  const float* c3_w  = (const float*)d_in[9];  const float* c3_b  = (const float*)d_in[10];
  const float* bn3_g = (const float*)d_in[11]; const float* bn3_b = (const float*)d_in[12];
  const float* c4_w  = (const float*)d_in[13]; const float* c4_b  = (const float*)d_in[14];
  const float* bn4_g = (const float*)d_in[15]; const float* bn4_b = (const float*)d_in[16];
  const float* efc_w = (const float*)d_in[17]; const float* efc_b = (const float*)d_in[18];
  const float* r_w   = (const float*)d_in[19]; const float* r_b   = (const float*)d_in[20];
  const float* e_w1  = (const float*)d_in[21]; const float* e_b1  = (const float*)d_in[22];
  const float* e_w2  = (const float*)d_in[23]; const float* e_b2  = (const float*)d_in[24];
  const float* dfc_w = (const float*)d_in[25]; const float* dfc_b = (const float*)d_in[26];
  const float* d1_w  = (const float*)d_in[27]; const float* d1_b  = (const float*)d_in[28];
  const float* bnd1_g= (const float*)d_in[29]; const float* bnd1_b= (const float*)d_in[30];
  const float* d2_w  = (const float*)d_in[31]; const float* d2_b  = (const float*)d_in[32];
  const float* bnd2_g= (const float*)d_in[33]; const float* bnd2_b= (const float*)d_in[34];
  const float* d3_w  = (const float*)d_in[35]; const float* d3_b  = (const float*)d_in[36];
  const float* bnd3_g= (const float*)d_in[37]; const float* bnd3_b= (const float*)d_in[38];
  const float* d4_w  = (const float*)d_in[39]; const float* d4_b  = (const float*)d_in[40];

  const int B = 512, L = 64, E = 64, H = 64;

  // ---- output layout: x_hat | gate | topk_idx ----
  float* out_xhat = (float*)d_out;                         // 512*3*64*64
  float* out_gate = out_xhat + (size_t)B * 3 * 64 * 64;    // 512*64
  int*   out_idx  = (int*)(out_gate + (size_t)B * E);      // 512*2

  // ---- workspace layout (floats) with liveness-based aliasing ----
  float* ws = (float*)d_ws;
  size_t off = 0;
  float* yb = ws + off; off += 16777216;            // pre-BN scratch (max: conv1 out)
  float* s1 = ws + off; off += 16777216;            // [512,32,32,32]
  float* regA = ws + off; off += 16777216;          // s2|s3|s4|d packed = 16.78M exactly
  float* s2   = regA;                               // [512,64,16,16]  8388608
  float* s3   = s2 + 8388608;                       // [512,128,8,8]   4194304
  float* s4   = s3 + 4194304;                       // [512,256,4,4]   2097152
  float* dbuf = s4 + 2097152;                       // [512,256,4,4]   2097152
  float* u1 = ws + off; off += 4194304;             // [512,128,8,8]
  float* u2 = ws + off; off += 8388608;             // [512,64,16,16]
  float* u3 = regA;                                 // aliases regA (dead by then)
  float* zb     = ws + off; off += 32768;           // [512,64]
  float* logits = ws + off; off += 32768;           // [512,64]
  float* zout   = ws + off; off += 32768;           // [512,64]
  float* topw   = ws + off; off += 1024;            // [512,2]
  float* ssbuf  = ws + off; off += 512;             // per-channel scale/shift

  const dim3 blk(128);   // 4 waves per block

  // ================= encoder =================
  { // conv1: 3 -> 32, 64x64 -> 32x32   (2 m-tiles)
    int N = B * 32 * 32;
    conv3x3s2_wmma<3><<<dim3(cdiv(N,16), cdiv(cdiv(32,16),4)), blk, 0, stream>>>(
        x, c1_w, c1_b, yb, B, 64, 64, 32, 32, 32);
    bn_stats<<<32, 256, 0, stream>>>(yb, bn1_g, bn1_b, ssbuf, B, 32, 1024);
    long tot = (long)B * 32 * 1024;
    bn_apply_relu<<<(unsigned)((tot + 255) / 256), 256, 0, stream>>>(yb, ssbuf, s1, 32, 1024, tot);
  }
  { // conv2: 32 -> 64, 32x32 -> 16x16  (4 m-tiles)
    int N = B * 16 * 16;
    conv3x3s2_wmma<32><<<dim3(cdiv(N,16), cdiv(cdiv(64,16),4)), blk, 0, stream>>>(
        s1, c2_w, c2_b, yb, B, 32, 32, 64, 16, 16);
    bn_stats<<<64, 256, 0, stream>>>(yb, bn2_g, bn2_b, ssbuf, B, 64, 256);
    long tot = (long)B * 64 * 256;
    bn_apply_relu<<<(unsigned)((tot + 255) / 256), 256, 0, stream>>>(yb, ssbuf, s2, 64, 256, tot);
  }
  { // conv3: 64 -> 128, 16x16 -> 8x8   (8 m-tiles)
    int N = B * 8 * 8;
    conv3x3s2_wmma<64><<<dim3(cdiv(N,16), cdiv(cdiv(128,16),4)), blk, 0, stream>>>(
        s2, c3_w, c3_b, yb, B, 16, 16, 128, 8, 8);
    bn_stats<<<128, 256, 0, stream>>>(yb, bn3_g, bn3_b, ssbuf, B, 128, 64);
    long tot = (long)B * 128 * 64;
    bn_apply_relu<<<(unsigned)((tot + 255) / 256), 256, 0, stream>>>(yb, ssbuf, s3, 128, 64, tot);
  }
  { // conv4: 128 -> 256, 8x8 -> 4x4    (16 m-tiles)
    int N = B * 4 * 4;
    conv3x3s2_wmma<128><<<dim3(cdiv(N,16), cdiv(cdiv(256,16),4)), blk, 0, stream>>>(
        s3, c4_w, c4_b, yb, B, 8, 8, 256, 4, 4);
    bn_stats<<<256, 256, 0, stream>>>(yb, bn4_g, bn4_b, ssbuf, B, 256, 16);
    long tot = (long)B * 256 * 16;
    bn_apply_relu<<<(unsigned)((tot + 255) / 256), 256, 0, stream>>>(yb, ssbuf, s4, 256, 16, tot);
  }
  // encoder FC: z[b,j], M=64, N=512, K=4096
  gemm_tt_wmma<<<dim3(cdiv(cdiv(B,16),4), cdiv(L,16)), blk, 0, stream>>>(
      efc_w, L, s4, 4096, efc_b, zb, 1, L, L, B, 4096);

  // ================= routing =================
  gemm_tt_wmma<<<dim3(cdiv(cdiv(B,16),4), cdiv(E,16)), blk, 0, stream>>>(
      r_w, E, zb, L, r_b, logits, 1, E, E, B, L);
  softmax_top2<<<cdiv(B, 256), 256, 0, stream>>>(logits, out_gate, out_idx, topw, B, E);
  expert_apply<<<B, 64, 0, stream>>>(zb, e_w1, e_b1, e_w2, e_b2, out_idx, topw, zout, L, H);

  // ================= decoder =================
  // decoder FC: M=4096, N=512, K=64
  gemm_tt_wmma<<<dim3(cdiv(cdiv(B,16),4), cdiv(4096,16)), blk, 0, stream>>>(
      dfc_w, 4096, zout, L, dfc_b, dbuf, 1, 4096, 4096, B, L);

  { // deconv1: [d|s4] 512ch, 4x4 -> 128ch 8x8   (8 m-tiles)
    int Npar = B * 4 * 4;
    deconv4x4s2_wmma<512><<<dim3(cdiv(Npar,16), cdiv(cdiv(128,16),4), 4), blk, 0, stream>>>(
        dbuf, s4, d1_w, d1_b, yb, B, 4, 4, 128, 8, 8, 0);
    bn_stats<<<128, 256, 0, stream>>>(yb, bnd1_g, bnd1_b, ssbuf, B, 128, 64);
    long tot = (long)B * 128 * 64;
    bn_apply_relu<<<(unsigned)((tot + 255) / 256), 256, 0, stream>>>(yb, ssbuf, u1, 128, 64, tot);
  }
  { // deconv2: [u1|s3] 256ch, 8x8 -> 64ch 16x16 (4 m-tiles)
    int Npar = B * 8 * 8;
    deconv4x4s2_wmma<256><<<dim3(cdiv(Npar,16), cdiv(cdiv(64,16),4), 4), blk, 0, stream>>>(
        u1, s3, d2_w, d2_b, yb, B, 8, 8, 64, 16, 16, 0);
    bn_stats<<<64, 256, 0, stream>>>(yb, bnd2_g, bnd2_b, ssbuf, B, 64, 256);
    long tot = (long)B * 64 * 256;
    bn_apply_relu<<<(unsigned)((tot + 255) / 256), 256, 0, stream>>>(yb, ssbuf, u2, 64, 256, tot);
  }
  { // deconv3: [u2|s2] 128ch, 16x16 -> 32ch 32x32 (2 m-tiles)
    int Npar = B * 16 * 16;
    deconv4x4s2_wmma<128><<<dim3(cdiv(Npar,16), cdiv(cdiv(32,16),4), 4), blk, 0, stream>>>(
        u2, s2, d3_w, d3_b, yb, B, 16, 16, 32, 32, 32, 0);
    bn_stats<<<32, 256, 0, stream>>>(yb, bnd3_g, bnd3_b, ssbuf, B, 32, 1024);
    long tot = (long)B * 32 * 1024;
    bn_apply_relu<<<(unsigned)((tot + 255) / 256), 256, 0, stream>>>(yb, ssbuf, u3, 32, 1024, tot);
  }
  { // deconv4: [u3|s1] 64ch, 32x32 -> 3ch 64x64, fused sigmoid -> x_hat (1 m-tile)
    int Npar = B * 32 * 32;
    deconv4x4s2_wmma<64><<<dim3(cdiv(Npar,16), 1, 4), blk, 0, stream>>>(
        u3, s1, d4_w, d4_b, out_xhat, B, 32, 32, 3, 64, 64, 1);
  }
}
